// LISTA_CPSS_7533372638002
// MI455X (gfx1250) — matrix operand order repaired and checked by reading, not compiled, not run on hardware
//
#include <hip/hip_runtime.h>
#include <hip/hip_bf16.h>

// ---------------------------------------------------------------------------
// LISTA-CPSS on MI455X (gfx1250): bf16 WMMA (f32 accumulate), LDS-resident x,
// bitwise top-k selection. Wave32 / WGP aware.
// T=16, N=128, M=512, B=16384, P=64.
// ---------------------------------------------------------------------------

typedef __attribute__((ext_vector_type(16))) __bf16 v16bf;
typedef __attribute__((ext_vector_type(8)))  float  v8f;

union FragAB {
    v16bf v;
    uint4 q[2];
    unsigned short u[16];
};

__device__ __forceinline__ unsigned short f2bfu(float f) {
    unsigned u = __float_as_uint(f);
    unsigned r = u + 0x7FFFu + ((u >> 16) & 1u);   // round-to-nearest-even
    return (unsigned short)(r >> 16);
}

// Packed f32x2 -> bf16x2 conversion (single VALU op).
__device__ __forceinline__ unsigned cvt_pk_bf16(float lo, float hi) {
    unsigned r;
    asm("v_cvt_pk_bf16_f32 %0, %1, %2" : "=v"(r) : "v"(lo), "v"(hi));
    return r;
}

// ---------------------------------------------------------------------------
// Prep: pack A (used as B-operand of u = x @ A^T) into WMMA B-fragment order.
// B element (lane, j): K = kt*32 + (lane>=16?16:0) + j  (K indexes m in 0..511)
//                      N = nt*16 + (lane&15)            (N indexes n in 0..127)
// value = A[N, K]  (A is [128, 512] row-major)
// frag = kt*8 + nt ; element offset = frag*512 + lane*16 + j  (== gid)
// ---------------------------------------------------------------------------
__global__ void lista_pack_A(const float* __restrict__ A,
                             unsigned short* __restrict__ Abf) {
    int gid  = blockIdx.x * 256 + threadIdx.x;     // 65536 total
    int j    = gid & 15;
    int lane = (gid >> 4) & 31;
    int frag = gid >> 9;                           // 0..127
    int nt   = frag & 7;
    int kt   = frag >> 3;
    int N    = nt * 16 + (lane & 15);              // 0..127
    int K    = kt * 32 + ((lane >> 4) << 4) + j;   // 0..511
    Abf[gid] = f2bfu(A[(size_t)N * 512 + K]);
}

// ---------------------------------------------------------------------------
// Prep: pack W[t] (B-operand of delta = r @ W_t) into WMMA B-fragment order.
// K indexes n in 0..127, N indexes m-col in 0..511. value = W[t, K, N].
// frag = (t*4 + kt)*32 + nt
// ---------------------------------------------------------------------------
__global__ void lista_pack_W(const float* __restrict__ W,
                             unsigned short* __restrict__ Wbf) {
    int gid  = blockIdx.x * 256 + threadIdx.x;     // 1048576 total
    int j    = gid & 15;
    int lane = (gid >> 4) & 31;
    int frag = gid >> 9;                           // 0..2047
    int nt   = frag & 31;
    int kt   = (frag >> 5) & 3;
    int t    = frag >> 7;
    int N    = nt * 16 + (lane & 15);              // 0..511
    int K    = kt * 32 + ((lane >> 4) << 4) + j;   // 0..127
    Wbf[gid] = f2bfu(W[((size_t)t * 128 + K) * 512 + N]);
}

// ---------------------------------------------------------------------------
// Main kernel: 512 threads (16 waves), 32 batch rows per workgroup.
// LDS: x f32 [32][512] (64KB) | x bf16 [32][512] (32KB) | r bf16 [32][128] (8KB)
// ---------------------------------------------------------------------------
__global__ __launch_bounds__(512)
void lista_cpss_kernel(const float* __restrict__ y,
                       const unsigned short* __restrict__ Abf,
                       const unsigned short* __restrict__ Wbf,
                       const float* __restrict__ theta,
                       const int* __restrict__ pP,
                       float* __restrict__ out) {
    constexpr int NN = 128, MM = 512, ROWS = 32, THREADS = 512, TT = 16;

    extern __shared__ unsigned char smem[];
    float*          ldsX   = (float*)smem;                                  // 64KB
    unsigned short* ldsXbf = (unsigned short*)(smem + ROWS * MM * 4);       // 32KB
    unsigned short* ldsR   = (unsigned short*)(smem + ROWS * MM * 4 + ROWS * MM * 2); // 8KB

    const int tid  = threadIdx.x;
    const int lane = tid & 31;
    const int w    = tid >> 5;          // wave id 0..15
    const int hh   = (lane >> 4) & 1;   // lane half
    const int ll   = lane & 15;
    const int rowBase = blockIdx.x * ROWS;
    const int p = *pP;

    // Stage y as bf16 into ldsR (A-operand for iteration 0).
    for (int k = 0; k < (ROWS * NN) / THREADS; ++k) {   // 8x
        int idx = tid + k * THREADS;
        int r = idx >> 7, c = idx & 127;
        ldsR[idx] = f2bfu(y[(size_t)(rowBase + r) * NN + c]);
    }

    // Static step-1 assignment: one 16x16 tile of u per wave.
    const int mt1 = w & 1;        // batch-row tile (rows 0-15 / 16-31)
    const int nt1 = w >> 1;       // n-column tile 0..7 (128 = 8*16)
    // y fragment in C/D layout, kept in registers.
    float yfrag[8];
    #pragma unroll
    for (int i = 0; i < 8; ++i)
        yfrag[i] = y[(size_t)(rowBase + mt1 * 16 + hh * 8 + i) * NN + nt1 * 16 + ll];

    // Static step-3 assignment: 4 column tiles of x per wave.
    const int mt3 = w & 1;
    const int ng  = w >> 1;       // ntiles ng*4 .. ng*4+3

    __syncthreads();

    for (int t = 0; t < TT; ++t) {
        // Prefetch this iteration's W_t fragments (used in step 3) so they are
        // cache-staged while step 1's WMMA chain executes.
        #pragma unroll
        for (int ks = 0; ks < 4; ++ks) {
            #pragma unroll
            for (int q = 0; q < 4; ++q) {
                int frag = (t * 4 + ks) * 32 + (ng * 4 + q);
                __builtin_prefetch(Wbf + ((size_t)frag * 512 + lane * 16), 0, 0);
            }
        }

        if (t > 0) {
            // ---- step 1: u = x @ A^T ; r = y - u -> ldsR (bf16) ----
            v8f acc = {0.f, 0.f, 0.f, 0.f, 0.f, 0.f, 0.f, 0.f};
            #pragma unroll 4
            for (int ks = 0; ks < 16; ++ks) {          // K = 512 = 16*32
                FragAB a, b;
                int kb = ks * 32 + hh * 16;            // A-layout K contiguous per lane-half
                const uint4* pa = (const uint4*)(ldsXbf + (size_t)(mt1 * 16 + ll) * MM + kb);
                a.q[0] = pa[0];                        // K kb..kb+7
                a.q[1] = pa[1];                        // K kb+8..kb+15
                const uint4* pb = (const uint4*)(Abf + ((size_t)(ks * 8 + nt1) * 512 + lane * 16));
                b.q[0] = pb[0];
                b.q[1] = pb[1];
                acc = __builtin_amdgcn_wmma_f32_16x16x32_bf16(
                        false, a.v, false, b.v, (short)0, acc, false, false);
            }
            #pragma unroll
            for (int i = 0; i < 8; i += 2) {
                float r0 = yfrag[i]     - acc[i];
                float r1 = yfrag[i + 1] - acc[i + 1];
                unsigned pk = cvt_pk_bf16(r0, r1);
                size_t ix0 = (size_t)(mt1 * 16 + hh * 8 + i) * NN + nt1 * 16 + ll;
                ldsR[ix0]      = (unsigned short)(pk & 0xFFFFu);
                ldsR[ix0 + NN] = (unsigned short)(pk >> 16);
            }
            __syncthreads();
        }

        // ---- step 3: x = x + r @ W_t  (t==0: x = y @ W_0, C = 0) ----
        v8f acc3[4];
        #pragma unroll
        for (int q = 0; q < 4; ++q) {
            if (t == 0) {
                acc3[q] = (v8f){0.f, 0.f, 0.f, 0.f, 0.f, 0.f, 0.f, 0.f};
            } else {
                #pragma unroll
                for (int i = 0; i < 8; ++i)
                    acc3[q][i] = ldsX[(size_t)(mt3 * 16 + hh * 8 + i) * MM + (ng * 4 + q) * 16 + ll];
            }
        }
        #pragma unroll
        for (int ks = 0; ks < 4; ++ks) {               // K = 128 = 4*32
            FragAB a;
            int kb = ks * 32 + hh * 16;
            const uint4* pa = (const uint4*)(ldsR + (size_t)(mt3 * 16 + ll) * NN + kb);
            a.q[0] = pa[0];
            a.q[1] = pa[1];
            #pragma unroll
            for (int q = 0; q < 4; ++q) {
                FragAB b;
                int frag = (t * 4 + ks) * 32 + (ng * 4 + q);
                const uint4* pb = (const uint4*)(Wbf + ((size_t)frag * 512 + lane * 16));
                b.q[0] = pb[0];
                b.q[1] = pb[1];
                acc3[q] = __builtin_amdgcn_wmma_f32_16x16x32_bf16(
                            false, a.v, false, b.v, (short)0, acc3[q], false, false);
            }
        }
        #pragma unroll
        for (int q = 0; q < 4; ++q) {
            #pragma unroll
            for (int i = 0; i < 8; ++i)
                ldsX[(size_t)(mt3 * 16 + hh * 8 + i) * MM + (ng * 4 + q) * 16 + ll] = acc3[q][i];
        }
        __syncthreads();

        // ---- support selection: p-th largest |x| per row via bitwise search ----
        {
            const int row = 2 * w + hh;               // 16 waves * 2 rows = 32 rows
            float th = fabsf(theta[t]);
            float vals[32];
            unsigned ab[32];
            #pragma unroll
            for (int c = 0; c < 32; ++c) {
                float v = ldsX[(size_t)row * MM + c * 16 + ll];
                vals[c] = v;
                ab[c] = __float_as_uint(v) & 0x7FFFFFFFu;
            }
            // 15 probes (bits 30..16): threshold resolved to ~2^-7 relative,
            // consistent with the bf16 matmul path's precision.
            unsigned prefix = 0;
            for (int bit = 30; bit >= 16; --bit) {
                unsigned cand = prefix | (1u << bit);
                int cnt = 0;
                #pragma unroll
                for (int c = 0; c < 32; ++c) cnt += (ab[c] >= cand) ? 1 : 0;
                cnt += __shfl_xor(cnt, 1, 16);        // 16-lane (per-row) reduction
                cnt += __shfl_xor(cnt, 2, 16);
                cnt += __shfl_xor(cnt, 4, 16);
                cnt += __shfl_xor(cnt, 8, 16);
                if (cnt >= p) prefix = cand;
            }
            float tau = fmaxf(__uint_as_float(prefix), th);
            #pragma unroll
            for (int c = 0; c < 32; c += 2) {
                float x0  = vals[c];
                float x1  = vals[c + 1];
                float ax0 = __uint_as_float(ab[c]);
                float ax1 = __uint_as_float(ab[c + 1]);
                float sh0 = copysignf(fmaxf(ax0 - th, 0.0f), x0);  // softshrink(theta)
                float sh1 = copysignf(fmaxf(ax1 - th, 0.0f), x1);
                float o0  = (ax0 > tau) ? x0 : sh0;
                float o1  = (ax1 > tau) ? x1 : sh1;
                size_t ix0 = (size_t)row * MM + c * 16 + ll;
                size_t ix1 = ix0 + 16;
                ldsX[ix0] = o0;
                ldsX[ix1] = o1;
                unsigned pk = cvt_pk_bf16(o0, o1);
                ldsXbf[ix0] = (unsigned short)(pk & 0xFFFFu);
                ldsXbf[ix1] = (unsigned short)(pk >> 16);
            }
        }
        __syncthreads();
    }

    // ---- write x -> d_out ----
    for (int k = 0; k < (ROWS * MM) / THREADS; ++k) {  // 32x
        int idx = tid + k * THREADS;
        out[(size_t)(rowBase + (idx >> 9)) * MM + (idx & 511)] = ldsX[idx];
    }
}

// ---------------------------------------------------------------------------
extern "C" void kernel_launch(void* const* d_in, const int* in_sizes, int n_in,
                              void* d_out, int out_size, void* d_ws, size_t ws_size,
                              hipStream_t stream) {
    const float* y     = (const float*)d_in[0];   // [16384,128]
    const float* A     = (const float*)d_in[1];   // [128,512]
    const float* W     = (const float*)d_in[2];   // [16,128,512]
    const float* theta = (const float*)d_in[3];   // [16]
    const int*   p     = (const int*)d_in[4];     // scalar
    float*       out   = (float*)d_out;           // [16384,512]

    // Scratch layout: Abf 128KB @0, Wbf 2MB @128KB  (needs ~2.13MB of ws)
    unsigned short* Abf = (unsigned short*)d_ws;
    unsigned short* Wbf = (unsigned short*)((char*)d_ws + 131072);

    lista_pack_A<<<65536 / 256, 256, 0, stream>>>(A, Abf);
    lista_pack_W<<<1048576 / 256, 256, 0, stream>>>(W, Wbf);

    size_t lds = 32 * 512 * 4 + 32 * 512 * 2 + 32 * 128 * 2;   // 106496 bytes
    lista_cpss_kernel<<<16384 / 32, 512, lds, stream>>>(y, Abf, Wbf, theta, p, out);
}
